// TreeLSTM_22119081575029
// MI455X (gfx1250) — compile-verified, matrix operand order checked by reading, not run on hardware
//
#include <hip/hip_runtime.h>
#include <cstdint>
#include <cstddef>

typedef __bf16 bf16;
typedef __attribute__((ext_vector_type(16))) __bf16 v16bf;
typedef __attribute__((ext_vector_type(8)))  float  v8f;

#define N_LEAVES 32768
#define TREE_DEPTH 16
#define N_NODES (2 * N_LEAVES - 1)
#define HSZ 256
#define XSZ 300
#define FEAT 1024
#define NREG 49
#define NCLS 5

// padded sizes (K multiple of 64, N multiple of 16)
#define XP 320            // X padded
#define RP 64             // regions padded
#define CATK 1344         // FEAT + XP (concat [context, embeds] K)
#define SENTN 320         // sent output cols padded (cols >= 300 are exactly 0)
#define IOU3H 768

static __device__ __forceinline__ float sigf(float x) {
  return 1.0f / (1.0f + __expf(-x));
}

// ---------------------------------------------------------------------------
// Tiled WMMA GEMM:  C[M,N] = act( A[M,K] * B[N,K]^T + bias[N] )
// A,B bf16 row-major; Cf optional f32 out, Cb optional bf16 out.
// Block = 256 threads = 8 waves (2x4 wave grid).
// Block tile 32(M) x 128(N); K step 64; double-buffered LDS.
// Software pipeline: global loads for tile i+1 issued BEFORE the WMMA work of
// tile i; the register->LDS stores (and their s_wait_loadcnt) land after the
// 4 v_wmma ops, hiding HBM latency behind the matrix pipe.
// Requires K % 64 == 0 (all operands are padded accordingly).
// ---------------------------------------------------------------------------
__launch_bounds__(256)
__global__ void gemm_bf16_wmma(const bf16* __restrict__ A, int lda,
                               const bf16* __restrict__ B, int ldb,
                               float* __restrict__ Cf, int ldcf,
                               bf16* __restrict__ Cb, int ldcb,
                               const float* __restrict__ bias,
                               int M, int N, int K, int act) {
  __shared__ bf16 As[2][32][64];
  __shared__ bf16 Bs[2][128][64];

  const int t    = threadIdx.x;
  const int lane = t & 31;
  const int wave = t >> 5;       // 0..7
  const int wm   = wave >> 2;    // 0..1  (16-row band)
  const int wn   = wave & 3;     // 0..3  (32-col band)
  const int half = lane >> 4;    // 0..1
  const int l16  = lane & 15;

  const int m0 = blockIdx.y * 32;
  const int n0 = blockIdx.x * 128;

  // staging: A 32x64 (8 bf16/thread), B 128x64 (32 bf16/thread)
  const int arow = t >> 3;             // 0..31
  const int acol = (t & 7) * 8;        // 16B per thread
  const int brow = t >> 1;             // 0..127
  const int bcol = (t & 1) * 32;       // 64B per thread
  const bool aok = (m0 + arow) < M;
  const bool bok = (n0 + brow) < N;
  const bf16* aptr = A + (size_t)(m0 + arow) * lda + acol;
  const bf16* bptr = B + (size_t)(n0 + brow) * ldb + bcol;

  int4 areg;
  int4 breg[4];

  auto gload = [&](int k0) {           // issue global loads into registers
    areg = aok ? *(const int4*)(aptr + k0) : make_int4(0, 0, 0, 0);
#pragma unroll
    for (int i = 0; i < 4; ++i)
      breg[i] = bok ? *(const int4*)(bptr + k0 + i * 8) : make_int4(0, 0, 0, 0);
    if (aok) __builtin_prefetch(aptr + k0 + 64, 0, 1);
    if (bok) __builtin_prefetch(bptr + k0 + 64, 0, 1);
  };
  auto lstore = [&](int buf) {         // registers -> LDS
    *(int4*)&As[buf][arow][acol] = areg;
#pragma unroll
    for (int i = 0; i < 4; ++i)
      *(int4*)&Bs[buf][brow][bcol + i * 8] = breg[i];
  };

  v8f acc0 = {}, acc1 = {};
  gload(0);
  lstore(0);
  int buf = 0;

  for (int k0 = 0; k0 < K; k0 += 64) {
    const bool more = (k0 + 64) < K;
    if (more) gload(k0 + 64);          // in flight during this tile's WMMAs
    __syncthreads();                   // buf tile visible; prev readers done

    // fragment loads per ISA 16-bit layout:
    // lanes 0-15 hold K {c..c+7, c+16..c+23}; lanes 16-31 K {c+8.., c+24..}
    const bf16* ap  = &As[buf][wm * 16 + l16][0];
    const bf16* b0p = &Bs[buf][wn * 32 + l16][0];
    const bf16* b1p = &Bs[buf][wn * 32 + 16 + l16][0];
    v16bf a0, a1, b00, b01, b10, b11;
#pragma unroll
    for (int i = 0; i < 8; ++i) {
      a0[i]      = ap[half * 8 + i];
      a0[8 + i]  = ap[16 + half * 8 + i];
      a1[i]      = ap[32 + half * 8 + i];
      a1[8 + i]  = ap[48 + half * 8 + i];
      b00[i]     = b0p[half * 8 + i];
      b00[8 + i] = b0p[16 + half * 8 + i];
      b10[i]     = b0p[32 + half * 8 + i];
      b10[8 + i] = b0p[48 + half * 8 + i];
      b01[i]     = b1p[half * 8 + i];
      b01[8 + i] = b1p[16 + half * 8 + i];
      b11[i]     = b1p[32 + half * 8 + i];
      b11[8 + i] = b1p[48 + half * 8 + i];
    }

    acc0 = __builtin_amdgcn_wmma_f32_16x16x32_bf16(false, a0, false, b00,
                                                   (short)0, acc0, false, false);
    acc1 = __builtin_amdgcn_wmma_f32_16x16x32_bf16(false, a0, false, b01,
                                                   (short)0, acc1, false, false);
    acc0 = __builtin_amdgcn_wmma_f32_16x16x32_bf16(false, a1, false, b10,
                                                   (short)0, acc0, false, false);
    acc1 = __builtin_amdgcn_wmma_f32_16x16x32_bf16(false, a1, false, b11,
                                                   (short)0, acc1, false, false);

    if (more) lstore(buf ^ 1);         // s_wait_loadcnt lands here, post-WMMA
    buf ^= 1;
  }

  // epilogue: acc element r -> M = r (lanes 0-15) / r+8 (lanes 16-31)
  const int gn0 = n0 + wn * 32 + l16;
  const int gn1 = gn0 + 16;
#pragma unroll
  for (int r = 0; r < 8; ++r) {
    const int gm = m0 + wm * 16 + r + half * 8;
    if (gm < M) {
      if (gn0 < N) {
        float v = acc0[r];
        if (bias) v += bias[gn0];
        if (act == 1) v = tanhf(v);
        else if (act == 2) v = sigf(v);
        if (Cf) Cf[(size_t)gm * ldcf + gn0] = v;
        if (Cb) Cb[(size_t)gm * ldcb + gn0] = (bf16)v;
      }
      if (gn1 < N) {
        float v = acc1[r];
        if (bias) v += bias[gn1];
        if (act == 1) v = tanhf(v);
        else if (act == 2) v = sigf(v);
        if (Cf) Cf[(size_t)gm * ldcf + gn1] = v;
        if (Cb) Cb[(size_t)gm * ldcb + gn1] = (bf16)v;
      }
    }
  }
}

// ---------------------------------------------------------------------------
// Elementwise / small kernels
// ---------------------------------------------------------------------------
__global__ void k_zero_bf16(bf16* p, long long n) {
  long long i = (long long)blockIdx.x * blockDim.x + threadIdx.x;
  if (i < n) p[i] = (bf16)0.0f;
}

// dst[drows,dcols] = bf16(src[srows,scols]) zero-padded
__global__ void k_convert_pad(const float* __restrict__ src, int srows, int scols,
                              bf16* __restrict__ dst, int drows, int dcols) {
  long long i = (long long)blockIdx.x * blockDim.x + threadIdx.x;
  long long n = (long long)drows * dcols;
  if (i >= n) return;
  int r = (int)(i / dcols), c = (int)(i % dcols);
  float v = (r < srows && c < scols) ? src[(size_t)r * scols + c] : 0.0f;
  dst[i] = (bf16)v;
}

__global__ void k_pad_bias(const float* __restrict__ src, int n,
                           float* __restrict__ dst, int nd) {
  int i = blockIdx.x * blockDim.x + threadIdx.x;
  if (i < nd) dst[i] = (i < n) ? src[i] : 0.0f;
}

// img_in_f[RP,XP] = image[49,1024] @ attn_w_in[300,1024]^T (padded f32)
__global__ void k_img_in(const float* __restrict__ image,
                         const float* __restrict__ w_in,
                         float* __restrict__ out) {
  long long i = (long long)blockIdx.x * blockDim.x + threadIdx.x;
  if (i >= (long long)RP * XP) return;
  int r = (int)(i / XP), x = (int)(i % XP);
  float s = 0.0f;
  if (r < NREG && x < XSZ) {
    const float* im = image + (size_t)r * FEAT;
    const float* w  = w_in + (size_t)x * FEAT;
    for (int k = 0; k < FEAT; ++k) s += im[k] * w[k];
  }
  out[i] = s;
}

// imageT_b[FEAT,RP] = bf16(image^T), zero pad regions
__global__ void k_transpose_image(const float* __restrict__ image,
                                  bf16* __restrict__ dst) {
  long long i = (long long)blockIdx.x * blockDim.x + threadIdx.x;
  if (i >= (long long)FEAT * RP) return;
  int f = (int)(i / RP), r = (int)(i % RP);
  float v = (r < NREG) ? image[(size_t)r * FEAT + f] : 0.0f;
  dst[i] = (bf16)v;
}

// embeds (leaves) into catin cols [1024, 1324)
__global__ void k_gather_emb(const int* __restrict__ wordid,
                             const int* __restrict__ mask,
                             const float* __restrict__ emb,
                             bf16* __restrict__ catin) {
  long long i = (long long)blockIdx.x * blockDim.x + threadIdx.x;
  if (i >= (long long)N_LEAVES * XSZ) return;
  int n = (int)(i / XSZ), c = (int)(i % XSZ);
  int wid = wordid[n] * mask[n];
  catin[(size_t)n * CATK + FEAT + c] = (bf16)emb[(size_t)wid * XSZ + c];
}

// row softmax over 49 regions; write bf16 atten with zero pad cols
__global__ void k_softmax(const float* __restrict__ logits,
                          bf16* __restrict__ atten) {
  int n = blockIdx.x * blockDim.x + threadIdx.x;
  if (n >= N_LEAVES) return;
  const float* lr = logits + (size_t)n * RP;
  float m = lr[0];
  for (int r = 1; r < NREG; ++r) m = fmaxf(m, lr[r]);
  float e[NREG], s = 0.0f;
  for (int r = 0; r < NREG; ++r) { e[r] = __expf(lr[r] - m); s += e[r]; }
  float inv = 1.0f / s;
  bf16* ar = atten + (size_t)n * RP;
  for (int r = 0; r < NREG; ++r) ar[r] = (bf16)(e[r] * inv);
  for (int r = NREG; r < RP; ++r) ar[r] = (bf16)0.0f;
}

// leaves: apply_node with c_in = c0
__global__ void k_apply_leaf(const float* __restrict__ iou,
                             const float* __restrict__ b_iou,
                             const float* __restrict__ c0,
                             float* __restrict__ hf, float* __restrict__ cf,
                             bf16* __restrict__ hb) {
  long long i = (long long)blockIdx.x * blockDim.x + threadIdx.x;
  if (i >= (long long)N_LEAVES * HSZ) return;
  int n = (int)(i / HSZ), j = (int)(i % HSZ);
  const float* row = iou + (size_t)n * IOU3H;
  float ig = row[j]           + b_iou[j];
  float og = row[HSZ + j]     + b_iou[HSZ + j];
  float ug = row[2 * HSZ + j] + b_iou[2 * HSZ + j];
  float cn = sigf(ig) * tanhf(ug) + c0[(size_t)n * HSZ + j];
  float hv = sigf(og) * tanhf(cn);
  size_t o = (size_t)n * HSZ + j;
  hf[o] = hv; cf[o] = cn; hb[o] = (bf16)hv;
}

// internal levels: c_red = f0*c(2i) + f1*c(2i+1); apply_node
__global__ void k_apply_level(const float* __restrict__ f,
                              const float* __restrict__ iou,
                              const float* __restrict__ b_iou,
                              int off_prev, int off, int sz_new,
                              float* __restrict__ hf, float* __restrict__ cf,
                              bf16* __restrict__ hb) {
  long long idx = (long long)blockIdx.x * blockDim.x + threadIdx.x;
  if (idx >= (long long)sz_new * HSZ) return;
  int i = (int)(idx / HSZ), j = (int)(idx % HSZ);
  float f0 = f[(size_t)i * 2 * HSZ + j];
  float f1 = f[(size_t)i * 2 * HSZ + HSZ + j];
  float cl = cf[(size_t)(off_prev + 2 * i) * HSZ + j];
  float cr = cf[(size_t)(off_prev + 2 * i + 1) * HSZ + j];
  float cred = f0 * cl + f1 * cr;
  const float* row = iou + (size_t)i * IOU3H;
  float ig = row[j]           + b_iou[j];
  float og = row[HSZ + j]     + b_iou[HSZ + j];
  float ug = row[2 * HSZ + j] + b_iou[2 * HSZ + j];
  float cn = sigf(ig) * tanhf(ug) + cred;
  float hv = sigf(og) * tanhf(cn);
  size_t o = (size_t)(off + i) * HSZ + j;
  hf[o] = hv; cf[o] = cn; hb[o] = (bf16)hv;
}

// out[N_NODES,5] = h @ lin_w^T + lin_b
__global__ void k_classifier(const float* __restrict__ hf,
                             const float* __restrict__ lin_w,
                             const float* __restrict__ lin_b,
                             float* __restrict__ out) {
  long long i = (long long)blockIdx.x * blockDim.x + threadIdx.x;
  if (i >= (long long)N_NODES * NCLS) return;
  int n = (int)(i / NCLS), c = (int)(i % NCLS);
  const float* hr = hf + (size_t)n * HSZ;
  const float* wr = lin_w + (size_t)c * HSZ;
  float s = 0.0f;
  for (int k = 0; k < HSZ; ++k) s += hr[k] * wr[k];
  out[i] = s + lin_b[c];
}

// ---------------------------------------------------------------------------
static inline dim3 g1(long long n) { return dim3((unsigned)((n + 255) / 256)); }

extern "C" void kernel_launch(void* const* d_in, const int* in_sizes, int n_in,
                              void* d_out, int out_size, void* d_ws, size_t ws_size,
                              hipStream_t stream) {
  (void)in_sizes; (void)n_in; (void)out_size; (void)ws_size;
  const int*   wordid     = (const int*)d_in[0];
  const int*   mask       = (const int*)d_in[1];
  const float* image      = (const float*)d_in[2];
  const float* c0         = (const float*)d_in[4];
  const float* emb        = (const float*)d_in[5];
  const float* attn_w_in  = (const float*)d_in[6];
  const float* attn_w_out = (const float*)d_in[7];
  const float* attn_b_out = (const float*)d_in[8];
  const float* w_iou      = (const float*)d_in[9];
  const float* u_iou      = (const float*)d_in[10];
  const float* b_iou      = (const float*)d_in[11];
  const float* u_f_w      = (const float*)d_in[12];
  const float* u_f_b      = (const float*)d_in[13];
  const float* lin_w      = (const float*)d_in[14];
  const float* lin_b      = (const float*)d_in[15];
  float* out = (float*)d_out;

  uint8_t* ws = (uint8_t*)d_ws;
  size_t cur = 0;
  auto alloc = [&](size_t bytes) -> void* {
    cur = (cur + 255) & ~(size_t)255;
    void* p = ws + cur; cur += bytes; return p;
  };

  bf16*  catin    = (bf16*) alloc((size_t)N_LEAVES * CATK * 2);   // [ctx | emb | 0pad]
  float* logits   = (float*)alloc((size_t)N_LEAVES * RP * 4);
  bf16*  atten    = (bf16*) alloc((size_t)N_LEAVES * RP * 2);
  bf16*  sentb    = (bf16*) alloc((size_t)N_LEAVES * XP * 2);
  float* iouf     = (float*)alloc((size_t)N_LEAVES * IOU3H * 4);
  float* hf       = (float*)alloc((size_t)N_NODES * HSZ * 4);
  float* cf       = (float*)alloc((size_t)N_NODES * HSZ * 4);
  bf16*  hb       = (bf16*) alloc((size_t)N_NODES * HSZ * 2);
  float* ff       = (float*)alloc((size_t)(N_LEAVES / 2) * 2 * HSZ * 4);
  float* ioulvl   = (float*)alloc((size_t)(N_LEAVES / 2) * IOU3H * 4);
  bf16*  w_out_b  = (bf16*) alloc((size_t)SENTN * CATK * 2);
  bf16*  w_iou_b  = (bf16*) alloc((size_t)IOU3H * XP * 2);
  bf16*  u_iou_b  = (bf16*) alloc((size_t)IOU3H * 2 * HSZ * 2);
  bf16*  u_f_w_b  = (bf16*) alloc((size_t)2 * HSZ * 2 * HSZ * 2);
  float* img_in_f = (float*)alloc((size_t)RP * XP * 4);
  bf16*  img_in_b = (bf16*) alloc((size_t)RP * XP * 2);
  bf16*  imageT_b = (bf16*) alloc((size_t)FEAT * RP * 2);
  float* b_out_p  = (float*)alloc((size_t)XP * 4);

  auto gemm = [&](const bf16* A, int lda, const bf16* B, int ldb,
                  float* Cf, int ldcf, bf16* Cb, int ldcb,
                  const float* bias, int M, int N, int K, int act) {
    dim3 grid((N + 127) / 128, (M + 31) / 32);
    gemm_bf16_wmma<<<grid, 256, 0, stream>>>(A, lda, B, ldb, Cf, ldcf, Cb, ldcb,
                                             bias, M, N, K, act);
  };

  // 1. prep: zero concat buffer, convert/pad weights
  k_zero_bf16<<<g1((long long)N_LEAVES * CATK), 256, 0, stream>>>(catin, (long long)N_LEAVES * CATK);
  k_convert_pad<<<g1((long long)SENTN * CATK), 256, 0, stream>>>(attn_w_out, XSZ, XSZ + FEAT, w_out_b, SENTN, CATK);
  k_convert_pad<<<g1((long long)IOU3H * XP), 256, 0, stream>>>(w_iou, IOU3H, XSZ, w_iou_b, IOU3H, XP);
  k_convert_pad<<<g1((long long)IOU3H * 2 * HSZ), 256, 0, stream>>>(u_iou, IOU3H, 2 * HSZ, u_iou_b, IOU3H, 2 * HSZ);
  k_convert_pad<<<g1((long long)2 * HSZ * 2 * HSZ), 256, 0, stream>>>(u_f_w, 2 * HSZ, 2 * HSZ, u_f_w_b, 2 * HSZ, 2 * HSZ);
  k_pad_bias<<<1, XP, 0, stream>>>(attn_b_out, XSZ, b_out_p, XP);
  k_img_in<<<g1((long long)RP * XP), 256, 0, stream>>>(image, attn_w_in, img_in_f);
  k_convert_pad<<<g1((long long)RP * XP), 256, 0, stream>>>(img_in_f, RP, XP, img_in_b, RP, XP);
  k_transpose_image<<<g1((long long)FEAT * RP), 256, 0, stream>>>(image, imageT_b);

  // 2. attention pipeline over the 32768 leaves
  k_gather_emb<<<g1((long long)N_LEAVES * XSZ), 256, 0, stream>>>(wordid, mask, emb, catin);
  // logits = embeds @ img_in^T   (A = catin cols 1024.., lda=CATK)
  gemm(catin + FEAT, CATK, img_in_b, XP, logits, RP, nullptr, 0, nullptr,
       N_LEAVES, RP, XP, 0);
  k_softmax<<<g1(N_LEAVES), 256, 0, stream>>>(logits, atten);
  // context = atten @ image  -> catin cols 0..1023 (bf16)
  gemm(atten, RP, imageT_b, RP, nullptr, 0, catin, CATK, nullptr,
       N_LEAVES, FEAT, RP, 0);
  // sent = tanh(catin @ w_out^T + b)  -> bf16 (cols >= 300 exactly 0)
  gemm(catin, CATK, w_out_b, CATK, nullptr, 0, sentb, XP, b_out_p,
       N_LEAVES, SENTN, CATK, 1);
  // iou_x = sent @ w_iou^T  (leaves mask == 1)
  gemm(sentb, XP, w_iou_b, XP, iouf, IOU3H, nullptr, 0, nullptr,
       N_LEAVES, IOU3H, XP, 0);
  k_apply_leaf<<<g1((long long)N_LEAVES * HSZ), 256, 0, stream>>>(iouf, b_iou, c0, hf, cf, hb);

  // 3. topological tree propagation (15 dense levels)
  int off_prev = 0, sz = N_LEAVES, off = N_LEAVES;
  for (int lvl = 0; lvl < TREE_DEPTH - 1; ++lvl) {
    int m = sz >> 1;
    const bf16* hcat = hb + (size_t)off_prev * HSZ;   // [m, 512] row-major view
    gemm(hcat, 2 * HSZ, u_f_w_b, 2 * HSZ, ff, 2 * HSZ, nullptr, 0, u_f_b,
         m, 2 * HSZ, 2 * HSZ, 2 /*sigmoid*/);
    gemm(hcat, 2 * HSZ, u_iou_b, 2 * HSZ, ioulvl, IOU3H, nullptr, 0, nullptr,
         m, IOU3H, 2 * HSZ, 0);
    k_apply_level<<<g1((long long)m * HSZ), 256, 0, stream>>>(
        ff, ioulvl, b_iou, off_prev, off, m, hf, cf, hb);
    off_prev = off; sz = m; off += m;
  }

  // 4. classifier
  k_classifier<<<g1((long long)N_NODES * NCLS), 256, 0, stream>>>(hf, lin_w, lin_b, out);
}
